// Net_12532714570098
// MI455X (gfx1250) — matrix operand order repaired and checked by reading, not compiled, not run on hardware
//
#include <hip/hip_runtime.h>
#include <hip/hip_bf16.h>
#include <math.h>

// ---------------------------------------------------------------------------
// Types for WMMA fragments (CDNA5 / gfx1250, wave32)
// ---------------------------------------------------------------------------
typedef _Float16 half8  __attribute__((ext_vector_type(8)));
typedef _Float16 half16 __attribute__((ext_vector_type(16)));
typedef float    f32x8  __attribute__((ext_vector_type(8)));

#define N_BATCH 256
#define CIN     2048
#define PP      49
#define NP      (N_BATCH * PP)   // 12544 spatial rows
#define DIM     128
#define NCLASS  1000
#define NPADDED 1024             // 1000 padded to 64 WMMA N-tiles
#define KFC     4096

// ===========================================================================
// Kernel 1: per-(n,p) L2 normalization over 2048 channels.
// Output: xn[np][c] f16, row-major (WMMA GEMM-A layout, 16B-aligned chunks).
// ===========================================================================
__global__ __launch_bounds__(256) void k_norm(const float* __restrict__ x,
                                              _Float16* __restrict__ xn) {
  const int np = blockIdx.x;              // n*49 + p
  const int n  = np / PP;
  const int p  = np - n * PP;
  const int t  = threadIdx.x;
  __shared__ float red[256];

  const float* base = x + (size_t)n * CIN * PP + p;
  float vals[8];
  float ss = 0.0f;
#pragma unroll
  for (int u = 0; u < 8; ++u) {
    float v = base[(size_t)(t + u * 256) * PP];
    vals[u] = v;
    ss += v * v;
  }
  red[t] = ss;
  __syncthreads();
  for (int off = 128; off; off >>= 1) {
    if (t < off) red[t] += red[t + off];
    __syncthreads();
  }
  const float inv = 1.0f / fmaxf(sqrtf(red[0]), 1e-12f);
  _Float16* out = xn + (size_t)np * CIN;
#pragma unroll
  for (int u = 0; u < 8; ++u) out[t + u * 256] = (_Float16)(vals[u] * inv);
}

// ===========================================================================
// Kernel 2: spatial mean over 7x7 per (n,c) for pooled branch.
// ===========================================================================
__global__ __launch_bounds__(256) void k_cmean(const float* __restrict__ x,
                                               float* __restrict__ xmean) {
  const int idx = blockIdx.x * 256 + threadIdx.x;   // n*2048 + c
  const float* p = x + (size_t)idx * PP;
  float s = 0.0f;
#pragma unroll
  for (int i = 0; i < PP; ++i) s += p[i];
  xmean[idx] = s * (1.0f / 49.0f);
}

// ===========================================================================
// Kernel 3: pack a row-major f32 weight matrix W[nvalid][Ktot] into f16
// WMMA B-fragment order. For 16x16x32 f16 WMMA, B (32K x 16N):
//   lane l holds N = 16*jt + (l&15); half h (0..15) holds K = kt*32 + 16*(h>=8) + 8*(l>=16) + (h&7)
// Flat pack index: (((kt*(Ncols/16) + jt)*32 + lane)*16 + h)
// ===========================================================================
__global__ __launch_bounds__(256) void k_pack_b(const float* __restrict__ src,
                                                _Float16* __restrict__ dst,
                                                int Ktot, int Ncols, int nvalid) {
  const int flat = blockIdx.x * 256 + threadIdx.x;
  if (flat >= Ktot * Ncols) return;
  const int h    = flat & 15;
  const int lane = (flat >> 4) & 31;
  const int rest = flat >> 9;
  const int nt   = Ncols >> 4;
  const int jt   = rest % nt;
  const int kt   = rest / nt;
  const int k    = kt * 32 + ((h >> 3) << 4) + ((lane >> 4) << 3) + (h & 7);
  const int nn   = jt * 16 + (lane & 15);
  const float v  = (nn < nvalid) ? src[(size_t)nn * Ktot + k] : 0.0f;
  dst[flat] = (_Float16)v;
}

// ---------------------------------------------------------------------------
// A-fragment load: 16-bit A 16x32 per ISA: lanes 0-15 M=0..15 K groups {0..7,
// 16..23}; lanes 16-31 same M, K groups {8..15, 24..31}. Two aligned 16B loads.
// ---------------------------------------------------------------------------
__device__ __forceinline__ half16 load_a_frag(const _Float16* row_k0, int lane) {
  const _Float16* p = row_k0 + ((lane >> 4) << 3);
  half8 lo = *(const half8*)(p);
  half8 hi = *(const half8*)(p + 16);
  return __builtin_shufflevector(lo, hi, 0, 1, 2, 3, 4, 5, 6, 7,
                                 8, 9, 10, 11, 12, 13, 14, 15);
}

// ===========================================================================
// Kernel 4: WMMA GEMM1: y[12544,128] = xn[12544,2048] @ w11^T, fused
// +b11, BN(scale/shift), ReLU.
// One wave per block (wave32): 784 blocks spread across all WGPs; each wave
// computes a 16M x 128N strip with 8 f32 accumulators.
// ===========================================================================
__global__ __launch_bounds__(32) void k_gemm1(
    const _Float16* __restrict__ xn, const _Float16* __restrict__ w11p,
    const float* __restrict__ b11,
    const float* __restrict__ g2, const float* __restrict__ b2,
    const float* __restrict__ m2, const float* __restrict__ v2,
    float* __restrict__ xf) {
  const int lane   = threadIdx.x & 31;
  const int m_base = blockIdx.x * 16;
  const int mrow   = m_base + (lane & 15);

  f32x8 acc[8];
#pragma unroll
  for (int j = 0; j < 8; ++j)
#pragma unroll
    for (int r = 0; r < 8; ++r) acc[j][r] = 0.0f;

  const _Float16* arow = xn + (size_t)mrow * CIN;
  for (int k0 = 0; k0 < CIN; k0 += 32) {
    half16 a = load_a_frag(arow + k0, lane);
    const _Float16* bb = w11p + ((size_t)(k0 >> 5) * 8) * 512 + (size_t)lane * 16;
#pragma unroll
    for (int j = 0; j < 8; ++j) {
      half16 bf = *(const half16*)(bb + (size_t)j * 512);
      acc[j] = __builtin_amdgcn_wmma_f32_16x16x32_f16(
          false, a, false, bf, (short)0, acc[j], false, false);
    }
  }

  // C/D layout: VGPR r -> M = r (lanes 0-15) or r+8 (lanes 16-31); N = lane&15
  const int mshift = (lane >> 4) << 3;
#pragma unroll
  for (int j = 0; j < 8; ++j) {
    const int nc   = j * 16 + (lane & 15);
    const float s  = g2[nc] * rsqrtf(v2[nc] + 1e-5f);
    const float sh = b2[nc] - m2[nc] * s;
    const float bi = b11[nc];
#pragma unroll
    for (int r = 0; r < 8; ++r) {
      const int mm = m_base + r + mshift;
      float val = (acc[j][r] + bi) * s + sh;
      xf[(size_t)mm * DIM + nc] = fmaxf(val, 0.0f);
    }
  }
}

// ===========================================================================
// Kernel 5: one VLAD branch per block (n = blockIdx.x, branch = blockIdx.y).
// logits -> softmax(K) -> aggregate -> s (mean) -> intra L2 -> global L2
// (folded as scalar) -> 64-wide matvec with wl. Everything in LDS (~49 KB).
// ===========================================================================
__global__ __launch_bounds__(256) void k_vlad(
    const float* __restrict__ xf,
    const float* __restrict__ wd1, const float* __restrict__ bd1,
    const float* __restrict__ cent1, const float* __restrict__ wl1,
    const float* __restrict__ bl1,
    const float* __restrict__ wd2, const float* __restrict__ bd2,
    const float* __restrict__ cent2, const float* __restrict__ wl2,
    const float* __restrict__ bl2,
    const float* __restrict__ wd3, const float* __restrict__ bd3,
    const float* __restrict__ cent3, const float* __restrict__ wl3,
    const float* __restrict__ bl3,
    float* __restrict__ obuf, float* __restrict__ sbuf) {
  const int n  = blockIdx.x;
  const int br = blockIdx.y;
  const int t  = threadIdx.x;

  int K = 8;
  const float *wd = wd1, *bd = bd1, *cent = cent1, *wl = wl1, *bl = bl1;
  if (br == 1) { K = 16; wd = wd2; bd = bd2; cent = cent2; wl = wl2; bl = bl2; }
  else if (br == 2) { K = 32; wd = wd3; bd = bd3; cent = cent3; wl = wl3; bl = bl3; }

  __shared__ float xs[PP * DIM];   // 25088 B
  __shared__ float aw[32 * PP];    //  6272 B
  __shared__ float vl[32 * DIM];   // 16384 B
  __shared__ float red[256];
  __shared__ float invk[32];
  __shared__ float scal;

  for (int i = t; i < PP * DIM; i += 256)
    xs[i] = xf[(size_t)(n * PP) * DIM + i];
  __syncthreads();

  // logits[k][p] = wd[k] . xs[p] + bd[k]
  for (int i = t; i < K * PP; i += 256) {
    const int k = i / PP, p = i - k * PP;
    const float* w = wd + k * DIM;
    float acc = bd[k];
    for (int d = 0; d < DIM; ++d) acc += w[d] * xs[p * DIM + d];
    aw[k * PP + p] = acc;
  }
  __syncthreads();

  // softmax over K per spatial position p
  if (t < PP) {
    float mx = -1e30f;
    for (int k = 0; k < K; ++k) mx = fmaxf(mx, aw[k * PP + t]);
    float sum = 0.0f;
    for (int k = 0; k < K; ++k) {
      float e = expf(aw[k * PP + t] - mx);
      aw[k * PP + t] = e;
      sum += e;
    }
    const float inv = 1.0f / sum;
    for (int k = 0; k < K; ++k) aw[k * PP + t] *= inv;
  }
  __syncthreads();

  // column sums per k (for the centroid term)
  if (t < K) {
    float cs = 0.0f;
    for (int p = 0; p < PP; ++p) cs += aw[t * PP + p];
    invk[t] = cs;
  }
  __syncthreads();

  // vlad[k][d] = sum_p a[k][p]*xs[p][d] - cent[k][d]*colsum[k]
  for (int i = t; i < K * DIM; i += 256) {
    const int k = i >> 7, d = i & 127;
    float acc = 0.0f;
    for (int p = 0; p < PP; ++p) acc += aw[k * PP + p] * xs[p * DIM + d];
    vl[i] = acc - cent[i] * invk[k];
  }
  __syncthreads();

  // s = mean(vlad)
  {
    float part = 0.0f;
    for (int i = t; i < K * DIM; i += 256) part += vl[i];
    red[t] = part;
    __syncthreads();
    for (int off = 128; off; off >>= 1) {
      if (t < off) red[t] += red[t + off];
      __syncthreads();
    }
    if (t == 0) sbuf[n * 3 + br] = red[0] / (float)(K * DIM);
    __syncthreads();
  }

  // intra-normalize each cluster over d
  if (t < K) {
    float ss = 0.0f;
    for (int d = 0; d < DIM; ++d) { float x = vl[t * DIM + d]; ss += x * x; }
    invk[t] = 1.0f / fmaxf(sqrtf(ss), 1e-12f);
  }
  __syncthreads();
  for (int i = t; i < K * DIM; i += 256) vl[i] *= invk[i >> 7];
  __syncthreads();

  // global L2 of the flattened vector -> scalar fold
  {
    float part = 0.0f;
    for (int i = t; i < K * DIM; i += 256) { float x = vl[i]; part += x * x; }
    red[t] = part;
    __syncthreads();
    for (int off = 128; off; off >>= 1) {
      if (t < off) red[t] += red[t + off];
      __syncthreads();
    }
    if (t == 0) scal = 1.0f / fmaxf(sqrtf(red[0]), 1e-12f);
    __syncthreads();
  }

  // o[j] = (sum_i vl[i]*wl[j][i]) * scal + bl[j]; 4 partials per output
  {
    const int len = K * DIM;
    const int j = t & 63, q = t >> 6;
    const int chunk = len >> 2;
    const float* w  = wl + (size_t)j * len + q * chunk;
    const float* vv = vl + q * chunk;
    float part = 0.0f;
    for (int i = 0; i < chunk; ++i) part += vv[i] * w[i];
    red[t] = part;
    __syncthreads();
    if (t < 64) {
      float o = (red[t] + red[t + 64] + red[t + 128] + red[t + 192]) * scal + bl[t];
      obuf[((size_t)n * 3 + br) * 64 + t] = o;
    }
  }
}

// ===========================================================================
// Kernel 6: SE gate + x1, pooled branch x2 (+BN), outer product (+BN fold),
// emit f16 rows [n][4096] for GEMM2.
// ===========================================================================
__global__ __launch_bounds__(256) void k_combine(
    const float* __restrict__ obuf, const float* __restrict__ sbuf,
    const float* __restrict__ xmean,
    const float* __restrict__ wse1, const float* __restrict__ wse2,
    const float* __restrict__ wp, const float* __restrict__ bp,
    const float* __restrict__ bnpg, const float* __restrict__ bnpb,
    const float* __restrict__ bnpm, const float* __restrict__ bnpv,
    const float* __restrict__ bnfg, const float* __restrict__ bnfb,
    const float* __restrict__ bnfm, const float* __restrict__ bnfv,
    _Float16* __restrict__ xo16) {
  const int n = blockIdx.x, t = threadIdx.x;
  __shared__ float x1[64], x2[64];

  if (t < 64) {
    const float s1 = sbuf[n * 3 + 0], s2 = sbuf[n * 3 + 1], s3 = sbuf[n * 3 + 2];
    const float h  = fmaxf(s1 * wse1[0] + s2 * wse1[1] + s3 * wse1[2], 0.0f);
    const float c0 = 1.0f / (1.0f + expf(-h * wse2[0]));
    const float c1 = 1.0f / (1.0f + expf(-h * wse2[1]));
    const float c2 = 1.0f / (1.0f + expf(-h * wse2[2]));
    x1[t] = c0 * obuf[((size_t)n * 3 + 0) * 64 + t] +
            c1 * obuf[((size_t)n * 3 + 1) * 64 + t] +
            c2 * obuf[((size_t)n * 3 + 2) * 64 + t];
  } else if (t < 128) {
    const int i = t - 64;
    const float* xm = xmean + (size_t)n * CIN;
    const float* w  = wp + (size_t)i * CIN;
    float acc = bp[i];
    for (int c = 0; c < CIN; ++c) acc += xm[c] * w[c];
    const float s = bnpg[i] * rsqrtf(bnpv[i] + 1e-5f);
    x2[i] = acc * s + (bnpb[i] - bnpm[i] * s);
  }
  __syncthreads();

#pragma unroll
  for (int u = 0; u < 16; ++u) {
    const int idx = u * 256 + t;           // i*64 + j, i from x2, j from x1
    const int i = idx >> 6, j = idx & 63;
    const float s = bnfg[idx] * rsqrtf(bnfv[idx] + 1e-5f);
    const float val = x1[j] * x2[i] * s + (bnfb[idx] - bnfm[idx] * s);
    xo16[(size_t)n * KFC + idx] = (_Float16)val;
  }
}

// ===========================================================================
// Kernel 7: WMMA GEMM2: out[256,1000] = xo16[256,4096] @ wfc^T + bfc
// (N padded to 1024, masked on store). One wave per block: grid (16, 8),
// wave = 16M x 128N, spread across WGPs.
// ===========================================================================
__global__ __launch_bounds__(32) void k_gemm2(
    const _Float16* __restrict__ xo16, const _Float16* __restrict__ wfcp,
    const float* __restrict__ bfc, float* __restrict__ out) {
  const int lane   = threadIdx.x & 31;
  const int m_base = blockIdx.x * 16;
  const int n_base = blockIdx.y * 128;
  const int mrow   = m_base + (lane & 15);

  f32x8 acc[8];
#pragma unroll
  for (int j = 0; j < 8; ++j)
#pragma unroll
    for (int r = 0; r < 8; ++r) acc[j][r] = 0.0f;

  const _Float16* arow = xo16 + (size_t)mrow * KFC;
  for (int k0 = 0; k0 < KFC; k0 += 32) {
    half16 a = load_a_frag(arow + k0, lane);
    const _Float16* bb =
        wfcp + ((size_t)(k0 >> 5) * 64 + (size_t)(n_base >> 4)) * 512 +
        (size_t)lane * 16;
#pragma unroll
    for (int j = 0; j < 8; ++j) {
      half16 bf = *(const half16*)(bb + (size_t)j * 512);
      acc[j] = __builtin_amdgcn_wmma_f32_16x16x32_f16(
          false, a, false, bf, (short)0, acc[j], false, false);
    }
  }

  const int mshift = (lane >> 4) << 3;
#pragma unroll
  for (int j = 0; j < 8; ++j) {
    const int nn = n_base + j * 16 + (lane & 15);
    if (nn < NCLASS) {
      const float bi = bfc[nn];
#pragma unroll
      for (int r = 0; r < 8; ++r) {
        const int mm = m_base + r + mshift;
        out[(size_t)mm * NCLASS + nn] = acc[j][r] + bi;
      }
    }
  }
}

// ===========================================================================
// Host launcher
// ===========================================================================
extern "C" void kernel_launch(void* const* d_in, const int* in_sizes, int n_in,
                              void* d_out, int out_size, void* d_ws,
                              size_t ws_size, hipStream_t stream) {
  (void)in_sizes; (void)n_in; (void)out_size; (void)ws_size;

  const float* x_source = (const float*)d_in[0];
  const float* w11   = (const float*)d_in[1];
  const float* b11   = (const float*)d_in[2];
  const float* bn2_g = (const float*)d_in[3];
  const float* bn2_b = (const float*)d_in[4];
  const float* bn2_m = (const float*)d_in[5];
  const float* bn2_v = (const float*)d_in[6];
  const float* wd1 = (const float*)d_in[7];   const float* bd1 = (const float*)d_in[8];
  const float* wd2 = (const float*)d_in[9];   const float* bd2 = (const float*)d_in[10];
  const float* wd3 = (const float*)d_in[11];  const float* bd3 = (const float*)d_in[12];
  const float* cent1 = (const float*)d_in[13];
  const float* cent2 = (const float*)d_in[14];
  const float* cent3 = (const float*)d_in[15];
  const float* wl1 = (const float*)d_in[16];  const float* bl1 = (const float*)d_in[17];
  const float* wl2 = (const float*)d_in[18];  const float* bl2 = (const float*)d_in[19];
  const float* wl3 = (const float*)d_in[20];  const float* bl3 = (const float*)d_in[21];
  const float* wse1 = (const float*)d_in[22];
  const float* wse2 = (const float*)d_in[23];
  const float* wp = (const float*)d_in[24];   const float* bp = (const float*)d_in[25];
  const float* bnp_g = (const float*)d_in[26];
  const float* bnp_b = (const float*)d_in[27];
  const float* bnp_m = (const float*)d_in[28];
  const float* bnp_v = (const float*)d_in[29];
  const float* bnf_g = (const float*)d_in[30];
  const float* bnf_b = (const float*)d_in[31];
  const float* bnf_m = (const float*)d_in[32];
  const float* bnf_v = (const float*)d_in[33];
  const float* wfc = (const float*)d_in[34];
  const float* bfc = (const float*)d_in[35];

  // Workspace layout (all sizes already 256B multiples). Total ~71.1 MB.
  char* ws = (char*)d_ws;
  size_t off = 0;
  _Float16* xn    = (_Float16*)(ws + off); off += (size_t)NP * CIN * 2;        // 51.4 MB
  _Float16* w11p  = (_Float16*)(ws + off); off += (size_t)CIN * DIM * 2;       // 0.5 MB
  _Float16* wfcp  = (_Float16*)(ws + off); off += (size_t)KFC * NPADDED * 2;   // 8.0 MB
  float*    xf    = (float*)(ws + off);    off += (size_t)NP * DIM * 4;        // 6.4 MB
  float*    xmean = (float*)(ws + off);    off += (size_t)N_BATCH * CIN * 4;   // 2.0 MB
  float*    obuf  = (float*)(ws + off);    off += (size_t)N_BATCH * 3 * 64 * 4;
  float*    sbuf  = (float*)(ws + off);    off += (size_t)N_BATCH * 3 * 4;
  off = (off + 255) & ~(size_t)255;
  _Float16* xo16  = (_Float16*)(ws + off); off += (size_t)N_BATCH * KFC * 2;   // 2.0 MB

  k_norm<<<NP, 256, 0, stream>>>(x_source, xn);
  k_cmean<<<(N_BATCH * CIN) / 256, 256, 0, stream>>>(x_source, xmean);
  k_pack_b<<<(CIN * DIM) / 256, 256, 0, stream>>>(w11, w11p, CIN, DIM, DIM);
  k_pack_b<<<(KFC * NPADDED) / 256, 256, 0, stream>>>(wfc, wfcp, KFC, NPADDED, NCLASS);
  // One wave32 per block: 784 / 128 independent blocks spread across all WGPs.
  k_gemm1<<<NP / 16, 32, 0, stream>>>(xn, w11p, b11, bn2_g, bn2_b, bn2_m,
                                      bn2_v, xf);
  k_vlad<<<dim3(N_BATCH, 3), 256, 0, stream>>>(
      xf, wd1, bd1, cent1, wl1, bl1, wd2, bd2, cent2, wl2, bl2, wd3, bd3,
      cent3, wl3, bl3, obuf, sbuf);
  k_combine<<<N_BATCH, 256, 0, stream>>>(obuf, sbuf, xmean, wse1, wse2, wp, bp,
                                         bnp_g, bnp_b, bnp_m, bnp_v, bnf_g,
                                         bnf_b, bnf_m, bnf_v, xo16);
  k_gemm2<<<dim3(N_BATCH / 16, 8), 32, 0, stream>>>(xo16, wfcp, bfc,
                                                    (float*)d_out);
}
